// GLMAttention_35046933135900
// MI455X (gfx1250) — matrix operand order repaired and checked by reading, not compiled, not run on hardware
//
#include <hip/hip_runtime.h>
#include <stdint.h>

// ---------------- problem constants ----------------
#define B_    2
#define S_    2048
#define H_    4096
#define NH_   32
#define NKV_  2
#define HD_   128
#define NQKV_ 4608                       // 4096 q + 256 k + 256 v
#define SCALE_ 0.08838834764831845f      // 128^-0.5

// ---------------- WMMA types (CDNA5, wave32) ----------------
typedef __bf16 bf16_t;
typedef bf16_t bf16x16 __attribute__((ext_vector_type(16)));
typedef float  f32x8   __attribute__((ext_vector_type(8)));

union Frag16 { bf16x16 v; uint32_t u[8]; };

__device__ __forceinline__ unsigned short f2bf(float f) {
  uint32_t u = __builtin_bit_cast(uint32_t, f);
  u += 0x7FFFu + ((u >> 16) & 1u);            // round-to-nearest-even
  return (unsigned short)(u >> 16);
}
__device__ __forceinline__ float bf2f(unsigned short h) {
  uint32_t u = ((uint32_t)h) << 16;
  return __builtin_bit_cast(float, u);
}
__device__ __forceinline__ uint32_t packbf(float lo, float hi) {
  return (uint32_t)f2bf(lo) | ((uint32_t)f2bf(hi) << 16);
}
__device__ __forceinline__ f32x8 wmma_bf16(const Frag16& a, const Frag16& b, f32x8 c) {
  // D = A(16x32 bf16) * B(32x16 bf16) + C(16x16 f32)
  return __builtin_amdgcn_wmma_f32_16x16x32_bf16(false, a.v, false, b.v,
                                                 (short)0, c, false, false);
}

// ---- CDNA5 async global->LDS DMA (ASYNCcnt path, CDNA5 ISA) ----
// INST_OFFSET is added to BOTH the LDS and global addresses, so one VGPR pair
// + offset:16 covers a 32-byte per-lane copy.
__device__ __forceinline__ void async_copy_b128(uint32_t lds_byte_addr, const void* gptr) {
  asm volatile("global_load_async_to_lds_b128 %0, %1, off"
               :: "v"(lds_byte_addr), "v"(gptr) : "memory");
}
__device__ __forceinline__ void async_copy_b128_off16(uint32_t lds_byte_addr, const void* gptr) {
  asm volatile("global_load_async_to_lds_b128 %0, %1, off offset:16"
               :: "v"(lds_byte_addr), "v"(gptr) : "memory");
}
__device__ __forceinline__ void wait_async0() {
  asm volatile("s_wait_asynccnt 0x0" ::: "memory");
}
// Generic flat pointer to a __shared__ object has the LDS byte offset in its
// low 32 bits (aperture in the high bits) -> truncate for the VDST operand.
__device__ __forceinline__ uint32_t lds_addr32(const void* p) {
  return (uint32_t)(uintptr_t)p;
}

// ---------------- f32 -> bf16 convert ----------------
__global__ __launch_bounds__(256) void cvt_bf16_kernel(const float* __restrict__ in,
                                                       unsigned short* __restrict__ out,
                                                       int n) {
  int stride = gridDim.x * blockDim.x;
  for (int i = blockIdx.x * blockDim.x + threadIdx.x; i < n; i += stride)
    out[i] = f2bf(in[i]);
}

// ---------------- tiled bf16 WMMA GEMM ----------------
// C[M,N] = A[M,K] * B[K,N] (+bias).  mode 0: f32 out.  mode 1: QKV scatter to bf16.
// Block tile 128x128, 8 waves (4x2), K-step 32. Per wave: 2x4 C fragments.
// A tile staged with global_load_async_to_lds_b128 (DMA, no VGPR round-trip);
// B tile staged with regular loads because it is transposed on the LDS store.
__global__ __launch_bounds__(256) void gemm_bf16_kernel(
    const unsigned short* __restrict__ A, const unsigned short* __restrict__ Bw,
    int M, int N, int K, int mode, const float* __restrict__ bias,
    float* __restrict__ outf, unsigned short* __restrict__ qb,
    unsigned short* __restrict__ kb, unsigned short* __restrict__ vb) {
  __shared__ unsigned short As[128][40];   // row-major tile, 80B row stride (16B aligned)
  __shared__ unsigned short Bs[128][40];   // transposed tile: Bs[n][k]
  const int tid  = threadIdx.x;
  const int lane = tid & 31;
  const int wv   = tid >> 5;
  const int half = lane >> 4, l15 = lane & 15;
  const int wm = (wv >> 1) << 5;           // wave M offset: 0/32/64/96
  const int wn = (wv & 1) << 6;            // wave N offset: 0/64
  const int m0 = blockIdx.y << 7, n0 = blockIdx.x << 7;

  f32x8 acc[2][4];
#pragma unroll
  for (int i = 0; i < 2; ++i)
#pragma unroll
    for (int j = 0; j < 4; ++j)
#pragma unroll
      for (int r = 0; r < 8; ++r) acc[i][j][r] = 0.0f;

  const int ar = tid >> 1, ac = (tid & 1) << 4;      // A staging: row, col chunk
  const int bkr = tid >> 3, bc = (tid & 7) << 4;     // B staging: k-row, n chunk
  const uint32_t a_lds = lds_addr32(&As[ar][ac]);

  for (int kc = 0; kc < K; kc += 32) {
    __syncthreads();
    { // stage A 128x32 via async DMA straight into LDS
      const unsigned short* g = A + (size_t)(m0 + ar) * K + kc + ac;
      async_copy_b128(a_lds, g);
      async_copy_b128_off16(a_lds, g);
    }
    { // stage B 32x128, store transposed (n-major) so K-pairs are contiguous
      const unsigned short* src = Bw + (size_t)(kc + bkr) * N + n0 + bc;
      uint4 v0 = *(const uint4*)src, v1 = *(const uint4*)(src + 8);
      unsigned short tmp[16];
      *(uint4*)tmp = v0; *(uint4*)(tmp + 8) = v1;
#pragma unroll
      for (int j = 0; j < 16; ++j) Bs[bc + j][bkr] = tmp[j];
      if (kc + 32 < K)
        __builtin_prefetch((const void*)(Bw + (size_t)(kc + 32 + bkr) * N + n0 + bc), 0, 1);
    }
    wait_async0();            // wave-local async completion
    __syncthreads();          // publish LDS across all 8 waves

    Frag16 af[2], bfr[4];
#pragma unroll
    for (int i = 0; i < 2; ++i) {        // A frag: lane m=l15, K=half*16+2t(+1)
      const unsigned short* row = &As[wm + (i << 4) + l15][0];
#pragma unroll
      for (int t = 0; t < 8; ++t) {
        int k = (half << 4) + (t << 1);
        af[i].u[t] = *(const uint32_t*)(row + k);
      }
    }
#pragma unroll
    for (int j = 0; j < 4; ++j) {        // B frag: lane n=l15, K=half*16+2t(+1)
      const unsigned short* row = &Bs[wn + (j << 4) + l15][0] + (half << 4);
#pragma unroll
      for (int t = 0; t < 8; ++t)
        bfr[j].u[t] = *(const uint32_t*)(row + (t << 1));
    }
#pragma unroll
    for (int i = 0; i < 2; ++i)
#pragma unroll
      for (int j = 0; j < 4; ++j)
        acc[i][j] = wmma_bf16(af[i], bfr[j], acc[i][j]);
  }

  // epilogue: C layout lane holds col n=l15, rows M = r + 8*half
#pragma unroll
  for (int i = 0; i < 2; ++i) {
#pragma unroll
    for (int j = 0; j < 4; ++j) {
      int fn  = n0 + wn + (j << 4) + l15;
      int fmb = m0 + wm + (i << 4) + (half << 3);
      if (mode == 0) {
#pragma unroll
        for (int r = 0; r < 8; ++r)
          outf[(size_t)(fmb + r) * N + fn] = acc[i][j][r];
      } else {
        float bv = bias[fn];
#pragma unroll
        for (int r = 0; r < 8; ++r) {
          int m = fmb + r;
          int bb = m >> 11, s = m & (S_ - 1);
          unsigned short hv = f2bf(acc[i][j][r] + bv);
          if (fn < 4096) {                         // Q -> [b][h][s][d]
            int hd = fn >> 7, d = fn & 127;
            qb[(((size_t)bb * NH_ + hd) * S_ + s) * HD_ + d] = hv;
          } else if (fn < 4352) {                  // K -> [b][g][s][d]
            int g = (fn - 4096) >> 7, d = fn & 127;
            kb[(((size_t)bb * NKV_ + g) * S_ + s) * HD_ + d] = hv;
          } else {                                 // V -> [b][g][s][d]
            int g = (fn - 4352) >> 7, d = fn & 127;
            vb[(((size_t)bb * NKV_ + g) * S_ + s) * HD_ + d] = hv;
          }
        }
      }
    }
  }
}

// ---------------- interleaved RoPE (in place, bf16 buffer, f32 math) ----------------
__global__ __launch_bounds__(256) void rope_kernel(unsigned short* __restrict__ buf,
                                                   const int* __restrict__ pos, int total) {
  int idx = blockIdx.x * blockDim.x + threadIdx.x;
  if (idx >= total) return;
  int j  = idx & 31;                 // rotation pair index 0..31
  int s  = (idx >> 5) & (S_ - 1);
  int bh = idx >> 16;                // 32 pairs * 2048 seq = 65536 per (b,h)
  float p   = (float)pos[s];
  float inv = __powf(10000.0f, -(float)j * (1.0f / 32.0f));
  float fr  = p * inv;
  float cs = __cosf(fr), sn = __sinf(fr);
  size_t base = ((size_t)bh * S_ + s) * HD_ + (j << 1);
  float x1 = bf2f(buf[base]), x2 = bf2f(buf[base + 1]);
  buf[base]     = f2bf(x1 * cs - x2 * sn);
  buf[base + 1] = f2bf(x2 * cs + x1 * sn);
}

// ---------------- flash attention (causal, GQA 16:1) ----------------
// Block: 256 thr = 8 waves, one wave per head (8 of the 16 heads sharing a KV group).
// Each wave: 16 query rows. Scores computed transposed (M=key, N=query).
__global__ __launch_bounds__(256) void attn_kernel(
    const unsigned short* __restrict__ qb, const unsigned short* __restrict__ kb,
    const unsigned short* __restrict__ vb, unsigned short* __restrict__ ctx) {
  __shared__ unsigned short Kt[32][136];   // K tile  [t][d], 272B row stride (16B aligned)
  __shared__ unsigned short Vt[128][40];   // V tileT [d][t]
  const int tid  = threadIdx.x, lane = tid & 31, wv = tid >> 5;
  const int half = lane >> 4, l15 = lane & 15;
  const int qtile = blockIdx.x;            // 0..127
  const int hh    = blockIdx.y;            // 0..3 -> (g, head half)
  const int b     = blockIdx.z;
  const int g = hh >> 1;
  const int h = g * 16 + ((hh & 1) << 3) + wv;
  const int q0 = qtile << 4;
  const int qrow = q0 + l15;

  // Q fragments (B-side, K-dim = d), loaded straight from global (pairs contiguous)
  Frag16 qf[4];
  {
    const unsigned short* qbase = qb + (((size_t)b * NH_ + h) * S_ + qrow) * HD_;
#pragma unroll
    for (int kc = 0; kc < 4; ++kc)
#pragma unroll
      for (int t = 0; t < 8; ++t) {
        int d = (kc << 5) + (half << 4) + (t << 1);
        qf[kc].u[t] = *(const uint32_t*)(qbase + d);
      }
  }

  f32x8 o[8];                              // O^T fragments: M=d chunk, N=q
#pragma unroll
  for (int f = 0; f < 8; ++f)
#pragma unroll
    for (int r = 0; r < 8; ++r) o[f][r] = 0.0f;
  float mrun = -3.0e38f, lrun = 0.0f;

  const unsigned short* kbb = kb + ((size_t)b * NKV_ + g) * (size_t)S_ * HD_;
  const unsigned short* vbb = vb + ((size_t)b * NKV_ + g) * (size_t)S_ * HD_;
  const int nkb = (q0 + 16 + 31) >> 5;     // causal key-block count (32 keys/block)
  const int str = tid >> 3, stc = (tid & 7) << 4;
  const uint32_t k_lds = lds_addr32(&Kt[str][stc]);

  for (int kbi = 0; kbi < nkb; ++kbi) {
    const int t0 = kbi << 5;
    __syncthreads();
    { // stage K 32x128 via async DMA straight into LDS
      const unsigned short* g_ = kbb + (size_t)(t0 + str) * HD_ + stc;
      async_copy_b128(k_lds, g_);
      async_copy_b128_off16(k_lds, g_);
    }
    { // stage V transposed (must round-trip through VGPRs)
      const unsigned short* src = vbb + (size_t)(t0 + str) * HD_ + stc;
      unsigned short tmp[16];
      *(uint4*)tmp       = *(const uint4*)src;
      *(uint4*)(tmp + 8) = *(const uint4*)(src + 8);
#pragma unroll
      for (int j = 0; j < 16; ++j) Vt[stc + j][str] = tmp[j];
    }
    wait_async0();
    __syncthreads();

    // S^T = K * Q^T : two 16x16 fragments (t 0..15, t 16..31), K-dim 128 = 4 steps
    f32x8 s0, s1;
#pragma unroll
    for (int r = 0; r < 8; ++r) { s0[r] = 0.0f; s1[r] = 0.0f; }
#pragma unroll
    for (int kc = 0; kc < 4; ++kc) {
      Frag16 a0, a1;
#pragma unroll
      for (int t = 0; t < 8; ++t) {
        int d = (kc << 5) + (half << 4) + (t << 1);
        a0.u[t] = *(const uint32_t*)(&Kt[l15][d]);
        a1.u[t] = *(const uint32_t*)(&Kt[l15 + 16][d]);
      }
      s0 = wmma_bf16(a0, qf[kc], s0);
      s1 = wmma_bf16(a1, qf[kc], s1);
    }

    // scale + causal mask + online softmax (lane owns query column qrow)
    float p0[8], p1[8];
    float mloc = -3.0e38f;
#pragma unroll
    for (int r = 0; r < 8; ++r) {
      int t = t0 + r + (half << 3);
      float v0 = s0[r] * SCALE_; if (t > qrow)      v0 = -3.0e38f;
      float v1 = s1[r] * SCALE_; if (t + 16 > qrow) v1 = -3.0e38f;
      p0[r] = v0; p1[r] = v1;
      mloc = fmaxf(mloc, fmaxf(v0, v1));
    }
    mloc = fmaxf(mloc, __shfl_xor(mloc, 16, 32));
    float mnew = fmaxf(mrun, mloc);
    float corr = __expf(mrun - mnew);
    float ssum = 0.0f;
#pragma unroll
    for (int r = 0; r < 8; ++r) {
      p0[r] = __expf(p0[r] - mnew);
      p1[r] = __expf(p1[r] - mnew);
      ssum += p0[r] + p1[r];
    }
    ssum += __shfl_xor(ssum, 16, 32);
    lrun = lrun * corr + ssum;
    mrun = mnew;
#pragma unroll
    for (int f = 0; f < 8; ++f)
#pragma unroll
      for (int r = 0; r < 8; ++r) o[f][r] *= corr;

    // repack P^T (C layout) -> B fragment (K=t 0..31, n=q) via half-swap shuffles
    float p0x[8], p1x[8];
#pragma unroll
    for (int r = 0; r < 8; ++r) {
      p0x[r] = __shfl_xor(p0[r], 16, 32);
      p1x[r] = __shfl_xor(p1[r], 16, 32);
    }
    Frag16 pf;
#pragma unroll
    for (int t = 0; t < 8; ++t) {
      int j0 = t << 1, j1 = j0 + 1;
      float e0, e1;
      if (half == 0) {                       // lanes 0-15 need t 0..15 (frag s0 rows)
        e0 = (j0 < 8) ? p0[j0] : p0x[j0 - 8];
        e1 = (j1 < 8) ? p0[j1] : p0x[j1 - 8];
      } else {                               // lanes 16-31 need t 16..31 (frag s1 rows)
        e0 = (j0 < 8) ? p1x[j0] : p1[j0 - 8];
        e1 = (j1 < 8) ? p1x[j1] : p1[j1 - 8];
      }
      pf.u[t] = packbf(e0, e1);
    }

    // O^T += V^T * P^T : 8 d-chunks
#pragma unroll
    for (int dc = 0; dc < 8; ++dc) {
      Frag16 vf;
      const unsigned short* vrow = &Vt[(dc << 4) + l15][0];
#pragma unroll
      for (int t = 0; t < 8; ++t) {
        int tt = (half << 4) + (t << 1);
        vf.u[t] = *(const uint32_t*)(vrow + tt);
      }
      o[dc] = wmma_bf16(vf, pf, o[dc]);
    }
  }

  // normalize and write ctx[b][s][h*128 + d] (bf16, feeds output GEMM)
  float rl = 1.0f / lrun;
  unsigned short* cb = ctx + ((size_t)b * S_ + qrow) * H_ + h * HD_;
#pragma unroll
  for (int dc = 0; dc < 8; ++dc)
#pragma unroll
    for (int r = 0; r < 8; ++r) {
      int d = (dc << 4) + r + (half << 3);
      cb[d] = f2bf(o[dc][r] * rl);
    }
}

// ---------------- launcher ----------------
extern "C" void kernel_launch(void* const* d_in, const int* in_sizes, int n_in,
                              void* d_out, int out_size, void* d_ws, size_t ws_size,
                              hipStream_t stream) {
  const float* hs   = (const float*)d_in[0];
  const int*   pos  = (const int*)d_in[1];
  const float* wqkv = (const float*)d_in[2];
  const float* bqkv = (const float*)d_in[3];
  const float* wd   = (const float*)d_in[4];
  float* out = (float*)d_out;

  unsigned short* ws = (unsigned short*)d_ws;
  size_t off = 0;
  unsigned short* hs_bf  = ws + off; off += (size_t)B_ * S_ * H_;
  unsigned short* wq_bf  = ws + off; off += (size_t)H_ * NQKV_;
  unsigned short* wd_bf  = ws + off; off += (size_t)H_ * H_;
  unsigned short* q_bf   = ws + off; off += (size_t)B_ * NH_ * S_ * HD_;
  unsigned short* k_bf   = ws + off; off += (size_t)B_ * NKV_ * S_ * HD_;
  unsigned short* v_bf   = ws + off; off += (size_t)B_ * NKV_ * S_ * HD_;
  unsigned short* ctx_bf = ws + off; off += (size_t)B_ * S_ * H_;

  cvt_bf16_kernel<<<8192, 256, 0, stream>>>(hs,   hs_bf, B_ * S_ * H_);
  cvt_bf16_kernel<<<8192, 256, 0, stream>>>(wqkv, wq_bf, H_ * NQKV_);
  cvt_bf16_kernel<<<8192, 256, 0, stream>>>(wd,   wd_bf, H_ * H_);

  dim3 g1(NQKV_ / 128, (B_ * S_) / 128);
  gemm_bf16_kernel<<<g1, 256, 0, stream>>>(hs_bf, wq_bf, B_ * S_, NQKV_, H_,
                                           1, bqkv, nullptr, q_bf, k_bf, v_bf);

  rope_kernel<<<(B_ * NH_  * S_ * 32) / 256, 256, 0, stream>>>(q_bf, pos, B_ * NH_  * S_ * 32);
  rope_kernel<<<(B_ * NKV_ * S_ * 32) / 256, 256, 0, stream>>>(k_bf, pos, B_ * NKV_ * S_ * 32);

  attn_kernel<<<dim3(S_ / 16, 4, B_), 256, 0, stream>>>(q_bf, k_bf, v_bf, ctx_bf);

  dim3 g2(H_ / 128, (B_ * S_) / 128);
  gemm_bf16_kernel<<<g2, 256, 0, stream>>>(ctx_bf, wd_bf, B_ * S_, H_, H_,
                                           0, nullptr, out, nullptr, nullptr, nullptr);
}
